// SparseMoE_65721589563853
// MI455X (gfx1250) — matrix operand order repaired and checked by reading, not compiled, not run on hardware
//
#include <hip/hip_runtime.h>
#include <hip/hip_bf16.h>

#define TOK_N 8192
#define DIM_D 1024
#define DIM_H 4096
#define NEXP  8

#define BM 128
#define BN 256
#define BK 32
#define LDA_PAD 20   // u32 stride per A row of K-pairs (16 data + 4 pad) -> 80B, 16B aligned
#define LDB_PAD 20   // u32 stride per B column of K-pairs
#define TILE_U32 (BN * 16)   // one swizzled B tile = 16KB

typedef __attribute__((ext_vector_type(16))) __bf16 bf16x16;
typedef __attribute__((ext_vector_type(8)))  float  f32x8;
typedef int i32x4 __attribute__((vector_size(16)));   // matches builtin param type

union FragAB { uint4 u[2]; bf16x16 v; };

#if __has_builtin(__builtin_amdgcn_global_load_async_to_lds_b128) && \
    __has_builtin(__builtin_amdgcn_s_wait_asynccnt)
#define USE_ASYNC_LDS 1
#else
#define USE_ASYNC_LDS 0
#endif

#if __has_builtin(__builtin_amdgcn_cvt_pk_bf16_f32)
typedef __attribute__((ext_vector_type(2))) __bf16 bf16x2;
__device__ __forceinline__ unsigned pack_bf16(float a, float b) {
  bf16x2 r = __builtin_amdgcn_cvt_pk_bf16_f32(a, b);
  return __builtin_bit_cast(unsigned, r);
}
#else
__device__ __forceinline__ unsigned pack_bf16(float a, float b) {
  unsigned ua = __float_as_uint(a), ub = __float_as_uint(b);
  ua = (ua + 0x7FFFu + ((ua >> 16) & 1u)) >> 16;
  ub = (ub + 0x7FFFu + ((ub >> 16) & 1u)) >> 16;
  return ua | (ub << 16);
}
#endif
__device__ __forceinline__ unsigned short f2bf(float a) {
  return (unsigned short)(pack_bf16(a, 0.0f) & 0xffffu);
}
__device__ __forceinline__ float gelu_exact(float v) {
  return 0.5f * v * (1.0f + erff(v * 0.70710678118654752440f));
}

// stage one (A,B) K-tile into the given LDS buffers.
// A: 2 x 16B per thread, per-lane gathered addresses (aOff4 precomputed, clamped)
// B: 64B per thread from the pre-swizzled contiguous weight tile
__device__ __forceinline__ void stage_tile(
    const uint4* A4, const int aOff4[2], int kq, const unsigned* wt,
    unsigned* Abuf, unsigned* Bbuf, int tid) {
#if USE_ASYNC_LDS
#pragma unroll
  for (int i = 0; i < 2; ++i) {
    const int idx = tid + i * 256;
    const int m = idx >> 2, q = idx & 3;
    __builtin_amdgcn_global_load_async_to_lds_b128(
        (i32x4*)(A4 + aOff4[i] + kq),
        (i32x4*)&Abuf[m * LDA_PAD + q * 4], 0, 0);
  }
#pragma unroll
  for (int c2 = 0; c2 < 4; ++c2)
    __builtin_amdgcn_global_load_async_to_lds_b128(
        (i32x4*)(wt + tid * 16 + 4 * c2),
        (i32x4*)&Bbuf[tid * LDB_PAD + 4 * c2], 0, 0);
#else
#pragma unroll
  for (int i = 0; i < 2; ++i) {
    const int idx = tid + i * 256;
    const int m = idx >> 2, q = idx & 3;
    const uint4 v = A4[aOff4[i] + kq];
    *(uint4*)&Abuf[m * LDA_PAD + q * 4] = v;
  }
  const uint4* w4 = (const uint4*)(wt + tid * 16);
  uint4* bst = (uint4*)&Bbuf[tid * LDB_PAD];
#pragma unroll
  for (int c2 = 0; c2 < 4; ++c2) bst[c2] = w4[c2];
#endif
}

// ---------------- gating: one wave (32 lanes) per token -------------------
__global__ __launch_bounds__(256) void gate_kernel(
    const float* __restrict__ x, const float* __restrict__ Wg,
    const float* __restrict__ bg, int* __restrict__ eidx, float* __restrict__ ew) {
  const int lane = threadIdx.x & 31;
  const int t = blockIdx.x * 8 + (threadIdx.x >> 5);
  if (t >= TOK_N) return;
  float acc[NEXP];
#pragma unroll
  for (int e = 0; e < NEXP; ++e) acc[e] = 0.f;
  const float* xr = x + (size_t)t * DIM_D;
  for (int d = lane; d < DIM_D; d += 32) {
    float xv = xr[d];
    const float* wr = Wg + d * NEXP;
#pragma unroll
    for (int e = 0; e < NEXP; ++e) acc[e] += xv * wr[e];
  }
#pragma unroll
  for (int e = 0; e < NEXP; ++e)
#pragma unroll
    for (int s = 16; s >= 1; s >>= 1) acc[e] += __shfl_xor(acc[e], s, 32);
  if (lane == 0) {
    float v[NEXP];
#pragma unroll
    for (int e = 0; e < NEXP; ++e) v[e] = acc[e] + bg[e];
    int i0 = 0; float v0 = v[0];
#pragma unroll
    for (int e = 1; e < NEXP; ++e) if (v[e] > v0) { v0 = v[e]; i0 = e; }
    int i1 = -1; float v1 = -3.4e38f;
#pragma unroll
    for (int e = 0; e < NEXP; ++e) if (e != i0 && v[e] > v1) { v1 = v[e]; i1 = e; }
    float ex = __expf(v1 - v0);           // v1 <= v0, stable
    float w0 = 1.0f / (1.0f + ex);
    float w1 = ex * w0;
    eidx[2 * t]     = i0;  eidx[2 * t + 1] = i1;
    ew[2 * t]       = w0;  ew[2 * t + 1]   = w1;
  }
}

// ------- deterministic compaction: wave e scans all tokens in order -------
__global__ void compact_kernel(const int* __restrict__ eidx,
                               const float* __restrict__ ew,
                               int* __restrict__ counts,
                               int* __restrict__ perm,
                               float* __restrict__ pw) {
  const int wave = threadIdx.x >> 5;
  const int lane = threadIdx.x & 31;
  if (wave == NEXP) {                       // identity slot (shared FFN)
    for (int i = lane; i < TOK_N; i += 32) {
      perm[NEXP * TOK_N + i] = i;
      pw[NEXP * TOK_N + i] = 1.0f;
    }
    if (lane == 0) counts[NEXP] = TOK_N;
    return;
  }
  const int e = wave;
  int off = 0;
  for (int base = 0; base < TOK_N; base += 32) {
    const int t = base + lane;
    float w = -1.0f;
    const int a = eidx[2 * t], b = eidx[2 * t + 1];
    if (a == e) w = ew[2 * t];
    else if (b == e) w = ew[2 * t + 1];
    const unsigned mask = (unsigned)__ballot(w >= 0.0f);
    const int pos = off + __popc(mask & ((1u << lane) - 1u));
    if (w >= 0.0f) { perm[e * TOK_N + pos] = t; pw[e * TOK_N + pos] = w; }
    off += __popc(mask);
  }
  if (lane == 0) counts[e] = off;
}

// --------- fp32 -> packed bf16-pair conversion (contiguous pairs) ---------
__global__ __launch_bounds__(256) void conv_x_kernel(
    const float* __restrict__ x, unsigned* __restrict__ out, int totalPairs) {
  const int i = blockIdx.x * 256 + threadIdx.x;
  if (i < totalPairs) {
    const float2 v = ((const float2*)x)[i];
    out[i] = pack_bf16(v.x, v.y);
  }
}

// -- fp32 [K, N] -> per-(K-tile,N-tile) LDS-image bf16 pairs ---------------
// out layout: [K/BK][N/BN][n:BN][kpair:16]  (16KB per tile, contiguous)
__global__ __launch_bounds__(256) void conv_w_tile_kernel(
    const float* __restrict__ W, unsigned* __restrict__ out, int Ncols) {
  const int n0 = blockIdx.x * BN;
  const int k0 = blockIdx.y * BK;
  const int t = threadIdx.x;
  const float* wp = W + (size_t)k0 * Ncols + n0 + t;
  unsigned bv[16];
#pragma unroll
  for (int j = 0; j < 16; ++j)
    bv[j] = pack_bf16(wp[(size_t)(2 * j) * Ncols], wp[(size_t)(2 * j + 1) * Ncols]);
  unsigned* op = out + ((size_t)blockIdx.y * gridDim.x + blockIdx.x) * TILE_U32 + t * 16;
  uint4* o4 = (uint4*)op;
  o4[0] = make_uint4(bv[0], bv[1], bv[2], bv[3]);
  o4[1] = make_uint4(bv[4], bv[5], bv[6], bv[7]);
  o4[2] = make_uint4(bv[8], bv[9], bv[10], bv[11]);
  o4[3] = make_uint4(bv[12], bv[13], bv[14], bv[15]);
}

// ---------------- GEMM1: Hb = GELU( gather(Xpk) @ Wsw + bias ) ------------
// Double-buffered, async-staged pipeline: stage k+1 while computing k.
__global__ __launch_bounds__(256) void moe_gemm1(
    const unsigned* __restrict__ Xpk, const unsigned* __restrict__ Wsw,
    const float* __restrict__ bias, const int* __restrict__ permS,
    const int* __restrict__ countPtr, unsigned short* __restrict__ Hb,
    int Kdim, int Ncols) {
  __shared__ unsigned Alds[2][BM * LDA_PAD];
  __shared__ unsigned Blds[2][BN * LDB_PAD];

  const int cnt = *countPtr;
  const int m0 = blockIdx.y * BM;
  if (m0 >= cnt) return;                    // => cnt >= 1 below
  const int n0 = blockIdx.x * BN;

  const int tid = threadIdx.x;
  const int lane = tid & 31, wave = tid >> 5;
  const int wm = wave & 1, wn = wave >> 1;  // 2 x 4 wave tiling
  const int l16 = lane & 15, hi = lane >> 4;

  // clamp gather index (rows >= cnt are dropped in the epilogue; any
  // in-bounds data is fine, no zero-fill needed)
  const uint4* A4 = (const uint4*)Xpk;      // row stride Kdim/8 uint4
  int aOff4[2];
#pragma unroll
  for (int i = 0; i < 2; ++i) {
    const int idx = tid + i * 256;
    const int m = idx >> 2, q = idx & 3;
    int gm = m0 + m; if (gm >= cnt) gm = cnt - 1;
    aOff4[i] = permS[gm] * (Kdim >> 3) + q;
    (void)m;
  }

  f32x8 c[4][4];
#pragma unroll
  for (int i = 0; i < 4; ++i)
#pragma unroll
    for (int j = 0; j < 4; ++j)
#pragma unroll
      for (int q = 0; q < 8; ++q) c[i][j][q] = 0.0f;

  const size_t wstep = (size_t)gridDim.x * TILE_U32;  // next K-tile of same N-tile
  const unsigned* wt0 = Wsw + (size_t)blockIdx.x * TILE_U32;
  const int nK = Kdim / BK;

  stage_tile(A4, aOff4, 0, wt0, Alds[0], Blds[0], tid);   // prologue: tile 0

  for (int kt = 0; kt < nK; ++kt) {
    const int cur = kt & 1;
#if USE_ASYNC_LDS
    __builtin_amdgcn_s_wait_asynccnt(0);
#endif
    __syncthreads();
    if (kt + 1 < nK)
      stage_tile(A4, aOff4, (kt + 1) * (BK >> 3), wt0 + (size_t)(kt + 1) * wstep,
                 Alds[cur ^ 1], Blds[cur ^ 1], tid);
    if (kt + 2 < nK)
      __builtin_prefetch(wt0 + (size_t)(kt + 2) * wstep + tid * 16, 0, 1);

    FragAB a[4], b[4];
#pragma unroll
    for (int am = 0; am < 4; ++am) {
      const uint4* ar = (const uint4*)&Alds[cur][(wm * 64 + am * 16 + l16) * LDA_PAD];
      a[am].u[0] = ar[hi];          // K0-7 / K8-15
      a[am].u[1] = ar[2 + hi];      // K16-23 / K24-31
    }
#pragma unroll
    for (int bn = 0; bn < 4; ++bn) {
      const uint4* br = (const uint4*)&Blds[cur][(wn * 64 + bn * 16 + l16) * LDB_PAD];
      b[bn].u[0] = br[hi];          // K0-7 / K8-15
      b[bn].u[1] = br[2 + hi];      // K16-23 / K24-31
    }
#pragma unroll
    for (int am = 0; am < 4; ++am)
#pragma unroll
      for (int bn = 0; bn < 4; ++bn)
        c[am][bn] = __builtin_amdgcn_wmma_f32_16x16x32_bf16(
            false, a[am].v, false, b[bn].v, (short)0, c[am][bn], false, false);
  }

  // epilogue: bias + exact GELU + bf16 store (compacted rows)
  float bcol[4];
#pragma unroll
  for (int bn = 0; bn < 4; ++bn) bcol[bn] = bias[n0 + wn * 64 + bn * 16 + l16];
#pragma unroll
  for (int am = 0; am < 4; ++am)
#pragma unroll
    for (int r = 0; r < 8; ++r) {
      const int gm = m0 + wm * 64 + am * 16 + hi * 8 + r;
      if (gm < cnt) {
        unsigned short* hrow = Hb + (size_t)gm * Ncols;
#pragma unroll
        for (int bn = 0; bn < 4; ++bn) {
          const float v = gelu_exact(c[am][bn][r] + bcol[bn]);
          hrow[n0 + wn * 64 + bn * 16 + l16] = f2bf(v);
        }
      }
    }
}

// ------ GEMM2: out[perm[i]] (+)= pw[i] * ( Hb[i,:] @ Wsw + bias ) ---------
__global__ __launch_bounds__(256) void moe_gemm2(
    const unsigned short* __restrict__ Hb, const unsigned* __restrict__ Wsw,
    const float* __restrict__ bias, const int* __restrict__ permS,
    const float* __restrict__ pwS, const int* __restrict__ countPtr,
    float* __restrict__ Out, int Kdim, int Ncols, int accumulate) {
  __shared__ unsigned Alds[2][BM * LDA_PAD];
  __shared__ unsigned Blds[2][BN * LDB_PAD];

  const int cnt = *countPtr;
  const int m0 = blockIdx.y * BM;
  if (m0 >= cnt) return;
  const int n0 = blockIdx.x * BN;

  const int tid = threadIdx.x;
  const int lane = tid & 31, wave = tid >> 5;
  const int wm = wave & 1, wn = wave >> 1;
  const int l16 = lane & 15, hi = lane >> 4;

  const uint4* A4 = (const uint4*)Hb;       // row stride Kdim/8 uint4
  int aOff4[2];
#pragma unroll
  for (int i = 0; i < 2; ++i) {
    const int idx = tid + i * 256;
    const int m = idx >> 2, q = idx & 3;
    int gm = m0 + m; if (gm >= cnt) gm = cnt - 1;
    aOff4[i] = gm * (Kdim >> 3) + q;
    (void)m;
  }

  f32x8 c[4][4];
#pragma unroll
  for (int i = 0; i < 4; ++i)
#pragma unroll
    for (int j = 0; j < 4; ++j)
#pragma unroll
      for (int q = 0; q < 8; ++q) c[i][j][q] = 0.0f;

  const size_t wstep = (size_t)gridDim.x * TILE_U32;
  const unsigned* wt0 = Wsw + (size_t)blockIdx.x * TILE_U32;
  const int nK = Kdim / BK;

  stage_tile(A4, aOff4, 0, wt0, Alds[0], Blds[0], tid);

  for (int kt = 0; kt < nK; ++kt) {
    const int cur = kt & 1;
#if USE_ASYNC_LDS
    __builtin_amdgcn_s_wait_asynccnt(0);
#endif
    __syncthreads();
    if (kt + 1 < nK)
      stage_tile(A4, aOff4, (kt + 1) * (BK >> 3), wt0 + (size_t)(kt + 1) * wstep,
                 Alds[cur ^ 1], Blds[cur ^ 1], tid);
    if (kt + 2 < nK)
      __builtin_prefetch(wt0 + (size_t)(kt + 2) * wstep + tid * 16, 0, 1);

    FragAB a[4], b[4];
#pragma unroll
    for (int am = 0; am < 4; ++am) {
      const uint4* ar = (const uint4*)&Alds[cur][(wm * 64 + am * 16 + l16) * LDA_PAD];
      a[am].u[0] = ar[hi];
      a[am].u[1] = ar[2 + hi];
    }
#pragma unroll
    for (int bn = 0; bn < 4; ++bn) {
      const uint4* br = (const uint4*)&Blds[cur][(wn * 64 + bn * 16 + l16) * LDB_PAD];
      b[bn].u[0] = br[hi];
      b[bn].u[1] = br[2 + hi];
    }
#pragma unroll
    for (int am = 0; am < 4; ++am)
#pragma unroll
      for (int bn = 0; bn < 4; ++bn)
        c[am][bn] = __builtin_amdgcn_wmma_f32_16x16x32_bf16(
            false, a[am].v, false, b[bn].v, (short)0, c[am][bn], false, false);
  }

  float bcol[4];
#pragma unroll
  for (int bn = 0; bn < 4; ++bn) bcol[bn] = bias[n0 + wn * 64 + bn * 16 + l16];
#pragma unroll
  for (int am = 0; am < 4; ++am)
#pragma unroll
    for (int r = 0; r < 8; ++r) {
      const int gm = m0 + wm * 64 + am * 16 + hi * 8 + r;
      if (gm < cnt) {
        const int tr = permS[gm];
        const float wgt = pwS[gm];
        float* orow = Out + (size_t)tr * Ncols;
#pragma unroll
        for (int bn = 0; bn < 4; ++bn) {
          const int col = n0 + wn * 64 + bn * 16 + l16;
          const float v = (c[am][bn][r] + bcol[bn]) * wgt;
          if (accumulate) orow[col] += v;   // stream-serialized across slots
          else            orow[col] = v;    // shared slot initializes out
        }
      }
    }
}

extern "C" void kernel_launch(void* const* d_in, const int* in_sizes, int n_in,
                              void* d_out, int out_size, void* d_ws, size_t ws_size,
                              hipStream_t stream) {
  const float* x   = (const float*)d_in[0];
  const float* Wg  = (const float*)d_in[1];
  const float* bg  = (const float*)d_in[2];
  const float* W1  = (const float*)d_in[3];
  const float* b1  = (const float*)d_in[4];
  const float* W2  = (const float*)d_in[5];
  const float* b2  = (const float*)d_in[6];
  const float* Ws1 = (const float*)d_in[7];
  const float* bs1 = (const float*)d_in[8];
  const float* Ws2 = (const float*)d_in[9];
  const float* bs2 = (const float*)d_in[10];
  float* out = (float*)d_out;

  char* ws = (char*)d_ws;
  const size_t MiB = 1024 * 1024;
  int*   eidx   = (int*)(ws);                                  // 2N ints
  float* ew     = (float*)(ws + 64 * 1024);                    // 2N floats
  int*   counts = (int*)(ws + 128 * 1024);                     // 16 ints
  int*   perm   = (int*)(ws + 132 * 1024);                     // 9*N ints
  float* pw     = (float*)(ws + 132 * 1024 + 9 * TOK_N * 4);   // 9*N floats
  unsigned short* hb  = (unsigned short*)(ws + 1 * MiB);       // [N, H]  bf16  64 MiB
  unsigned*       xpk = (unsigned*)(ws + 65 * MiB);            // [N, D/2] u32  16 MiB
  unsigned*       w1p = (unsigned*)(ws + 81 * MiB);            // swizzled W1    8 MiB
  unsigned*       w2p = (unsigned*)(ws + 89 * MiB);            // swizzled W2    8 MiB

  gate_kernel<<<TOK_N / 8, 256, 0, stream>>>(x, Wg, bg, eidx, ew);
  compact_kernel<<<1, 32 * (NEXP + 1), 0, stream>>>(eidx, ew, counts, perm, pw);

  const int xPairs = TOK_N * DIM_D / 2;                        // 4M
  conv_x_kernel<<<xPairs / 256, 256, 0, stream>>>(x, xpk, xPairs);

  // shared slot (index NEXP) first: initializes `out`; experts then accumulate.
  const int order[NEXP + 1] = {NEXP, 0, 1, 2, 3, 4, 5, 6, 7};
  for (int si = 0; si < NEXP + 1; ++si) {
    const int s = order[si];
    const float* W1s = (s == NEXP) ? Ws1 : (W1 + (size_t)s * DIM_D * DIM_H);
    const float* b1s = (s == NEXP) ? bs1 : (b1 + (size_t)s * DIM_H);
    const float* W2s = (s == NEXP) ? Ws2 : (W2 + (size_t)s * DIM_H * DIM_D);
    const float* b2s = (s == NEXP) ? bs2 : (b2 + (size_t)s * DIM_D);

    dim3 c1(DIM_H / BN, DIM_D / BK);   // 16 x 32
    conv_w_tile_kernel<<<c1, 256, 0, stream>>>(W1s, w1p, DIM_H);
    dim3 c2(DIM_D / BN, DIM_H / BK);   // 4 x 128
    conv_w_tile_kernel<<<c2, 256, 0, stream>>>(W2s, w2p, DIM_D);

    dim3 g1(DIM_H / BN, TOK_N / BM);   // 16 x 64, early-exit past count
    moe_gemm1<<<g1, 256, 0, stream>>>(xpk, w1p, b1s, perm + s * TOK_N,
                                      counts + s, hb, DIM_D, DIM_H);
    dim3 g2(DIM_D / BN, TOK_N / BM);   // 4 x 64
    moe_gemm2<<<g2, 256, 0, stream>>>(hb, w2p, b2s, perm + s * TOK_N,
                                      pw + s * TOK_N, counts + s, out,
                                      DIM_H, DIM_D, (s == NEXP) ? 0 : 1);
  }
}
